// SimpleNetW_2542620639564
// MI455X (gfx1250) — compile-verified, matrix operand order checked by reading, not run on hardware
//
#include <hip/hip_runtime.h>
#include <hip/hip_bf16.h>
#include <stdint.h>

// ---------------------------------------------------------------------------
// GCN forward for MI455X (gfx1250, wave32).
// Dense GEMMs -> v_wmma_f32_16x16x32_bf16 (fp32 accumulate).
// Edge scatter -> native global_atomic_add_f32 (unsafeAtomicAdd).
// ---------------------------------------------------------------------------

typedef __bf16 bf16x16 __attribute__((ext_vector_type(16)));
typedef float  f32x8   __attribute__((ext_vector_type(8)));

struct alignas(16) I4 { int a, b, c, d; };
union ABOp { I4 i[2]; bf16x16 v; };
struct alignas(8) F2 { float x, y; };

#define NEG_INF_F (-1e30f)

__device__ __forceinline__ uint16_t f2bf(float f) {
    uint32_t u = __float_as_uint(f);
    uint32_t r = u + 0x7FFFu + ((u >> 16) & 1u);   // round-to-nearest-even
    return (uint16_t)(r >> 16);
}

__device__ __forceinline__ void atomAddF32(float* p, float v) {
    unsafeAtomicAdd(p, v);   // native global_atomic_add_f32 on CDNA5
}

// -------------------------- small prep kernels -----------------------------

__global__ void k_convert_bf16(const float* __restrict__ src,
                               uint16_t* __restrict__ dst, int n) {
    int i = blockIdx.x * blockDim.x + threadIdx.x;
    if (i < n) dst[i] = f2bf(src[i]);
}

// deg = 1 (self loop), zero xsum
__global__ void k_init(float* __restrict__ deg, float* __restrict__ xsum, int n) {
    int i = blockIdx.x * blockDim.x + threadIdx.x;
    if (i < n) deg[i] = 1.0f;
    if (i < 64) xsum[i] = 0.0f;
}

__global__ void k_deg_edges(const int* __restrict__ row,
                            float* __restrict__ deg, int e) {
    int i = blockIdx.x * blockDim.x + threadIdx.x;
    if (i < e) atomAddF32(&deg[row[i]], 1.0f);
}

__global__ void k_rsqrt_inplace(float* __restrict__ deg, int n) {
    int i = blockIdx.x * blockDim.x + threadIdx.x;
    if (i < n) deg[i] = rsqrtf(deg[i]);       // deg >= 1 always (self loop)
}

// WT[n*K + k] = bf16(W[k*64 + n])   (W is [K x 64] row-major fp32)
__global__ void k_prep_wt(const float* __restrict__ W,
                          uint16_t* __restrict__ WT, int K) {
    int i = blockIdx.x * blockDim.x + threadIdx.x;
    if (i < K * 64) {
        int k = i >> 6, n = i & 63;
        WT[n * K + k] = f2bf(W[i]);
    }
}

// ------------------------------ WMMA GEMM ----------------------------------
// C[nrows x 64] = A[nrows x K](bf16) * W[K x 64], with WT = W^T ([64 x K] bf16).
// One wave -> 16 rows x 64 cols (4 accumulators), K-loop step 32.
template <int K>
__global__ __launch_bounds__(128)
void k_gemm(const uint16_t* __restrict__ A, const uint16_t* __restrict__ WT,
            float* __restrict__ C, int nrows) {
    __shared__ uint16_t lw[64 * K];

    // cooperative copy of W^T into LDS (I4 = 8 bf16 per transfer)
    {
        I4* d = (I4*)lw;
        const I4* s = (const I4*)WT;
        const int cnt = 64 * K / 8;
        for (int i = threadIdx.x; i < cnt; i += 128) d[i] = s[i];
    }
    __syncthreads();

    const int wid  = threadIdx.x >> 5;
    const long m0  = ((long)blockIdx.x * 4 + wid) * 16;
    if (m0 >= nrows) return;                      // wave-uniform branch

    const int lane = threadIdx.x & 31;
    const int lm   = lane & 15;
    const int kb   = (lane < 16) ? 0 : 8;         // ISA 16-bit A-operand layout
    const int rb   = (lane < 16) ? 0 : 8;         // f32 C/D layout row base

    const uint16_t* arow = A + (size_t)(m0 + lm) * K;

    f32x8 acc[4] = {};
#pragma unroll
    for (int kk = 0; kk < K; kk += 32) {
        ABOp ua;
        ua.i[0] = *(const I4*)(arow + kk + kb);
        ua.i[1] = *(const I4*)(arow + kk + 16 + kb);
        const bf16x16 a = ua.v;
#pragma unroll
        for (int nt = 0; nt < 4; ++nt) {
            const uint16_t* brow = lw + (size_t)(nt * 16 + lm) * K;
            ABOp ub;
            ub.i[0] = *(const I4*)(brow + kk + kb);      // ds_load_b128
            ub.i[1] = *(const I4*)(brow + kk + 16 + kb);
            acc[nt] = __builtin_amdgcn_wmma_f32_16x16x32_bf16(
                false, a, false, ub.v, (short)0, acc[nt], false, false);
        }
    }
#pragma unroll
    for (int nt = 0; nt < 4; ++nt)
#pragma unroll
        for (int r = 0; r < 8; ++r)
            C[(size_t)(m0 + rb + r) * 64 + nt * 16 + lm] = acc[nt][r];
}

// --------------------------- message passing -------------------------------

// agg[i] = dinv[node]^2 * t[i]   (self-loop term; also (re)initializes agg)
__global__ void k_selfinit(const float* __restrict__ dinv,
                           const float* __restrict__ t,
                           float* __restrict__ agg, int n64) {
    int i = blockIdx.x * blockDim.x + threadIdx.x;
    if (i < n64) {
        float d = dinv[i >> 6];
        agg[i] = d * d * t[i];
    }
}

// one wave per edge: agg[row][:] += dinv[row]*dinv[col] * t[col][:]
__global__ __launch_bounds__(256)
void k_edge_agg(const int* __restrict__ row, const int* __restrict__ col,
                const float* __restrict__ dinv, const float* __restrict__ t,
                float* __restrict__ agg, int e) {
    int wave = blockIdx.x * 8 + (threadIdx.x >> 5);
    if (wave >= e) return;                        // wave-uniform
    int lane = threadIdx.x & 31;
    __builtin_prefetch(row + wave + 4096, 0, 0);  // global_prefetch_b8
    int r = row[wave], c = col[wave];
    float nrm = dinv[r] * dinv[c];
    F2 v = *(const F2*)(t + (size_t)c * 64 + lane * 2);
    float* dst = agg + (size_t)r * 64 + lane * 2;
    atomAddF32(dst,     v.x * nrm);
    atomAddF32(dst + 1, v.y * nrm);
}

// h = relu(agg + b), also emit bf16 copy for the next GEMM
__global__ void k_relu_bias(const float* __restrict__ agg,
                            const float* __restrict__ b,
                            float* __restrict__ h,
                            uint16_t* __restrict__ hb, int n64) {
    int i = blockIdx.x * blockDim.x + threadIdx.x;
    if (i < n64) {
        float v = fmaxf(agg[i] + b[i & 63], 0.0f);
        h[i] = v;
        hb[i] = f2bf(v);
    }
}

// ------------------------------ head kernels -------------------------------

// t5[n] = h[n,:] . W5
__global__ void k_dot64(const float* __restrict__ h, const float* __restrict__ W5,
                        float* __restrict__ t5, int n) {
    int i = blockIdx.x * blockDim.x + threadIdx.x;
    if (i >= n) return;
    const float* r = h + (size_t)i * 64;
    float s = 0.0f;
#pragma unroll
    for (int k = 0; k < 64; ++k) s = fmaf(r[k], W5[k], s);
    t5[i] = s;
}

__global__ void k_selfinit_s(const float* __restrict__ dinv,
                             const float* __restrict__ t5,
                             float* __restrict__ agg5, int n) {
    int i = blockIdx.x * blockDim.x + threadIdx.x;
    if (i < n) { float d = dinv[i]; agg5[i] = d * d * t5[i]; }
}

__global__ void k_edge_agg_s(const int* __restrict__ row, const int* __restrict__ col,
                             const float* __restrict__ dinv,
                             const float* __restrict__ t5,
                             float* __restrict__ agg5, int e) {
    int i = blockIdx.x * blockDim.x + threadIdx.x;
    if (i < e) {
        int r = row[i], c = col[i];
        atomAddF32(&agg5[r], dinv[r] * dinv[c] * t5[c]);
    }
}

// feature-parallel partial sums of h (for x_mean); block = 64 threads, 512 nodes
__global__ __launch_bounds__(64)
void k_xmean_part(const float* __restrict__ h, float* __restrict__ xsum, int n) {
    int f = threadIdx.x;
    int n0 = blockIdx.x * 512;
    int n1 = min(n0 + 512, n);
    float s = 0.0f;
    for (int nn = n0; nn < n1; ++nn) s += h[(size_t)nn * 64 + f];
    atomAddF32(&xsum[f], s);
}

// x_mean head: v -> out[N+1], prob_nothing -> scal[2]
__global__ __launch_bounds__(64)
void k_head(const float* __restrict__ xsum,
            const float* __restrict__ Wv,  const float* __restrict__ bv,
            const float* __restrict__ Wdn, const float* __restrict__ bdn,
            float* __restrict__ out, float* __restrict__ scal, int n) {
    __shared__ float xm[64];
    xm[threadIdx.x] = xsum[threadIdx.x] / (float)n;
    __syncthreads();
    if (threadIdx.x == 0) {
        float v = bv[0], pn = bdn[0];
        for (int k = 0; k < 64; ++k) {
            v  = fmaf(xm[k], Wv[k],  v);
            pn = fmaf(xm[k], Wdn[k], pn);
        }
        out[n + 1] = v;
        scal[2] = pn;
    }
}

// ------------------------------ masked softmax -----------------------------

__global__ __launch_bounds__(256)
void k_max_part(const float* __restrict__ agg5, const float* __restrict__ b5,
                const int* __restrict__ ready, float* __restrict__ pmax, int n) {
    __shared__ float s[256];
    int i = blockIdx.x * 256 + threadIdx.x;
    float v = NEG_INF_F;
    if (i < n && ready[i] != 0) v = agg5[i] + b5[0];
    s[threadIdx.x] = v;
    __syncthreads();
    for (int st = 128; st > 0; st >>= 1) {
        if (threadIdx.x < st) s[threadIdx.x] = fmaxf(s[threadIdx.x], s[threadIdx.x + st]);
        __syncthreads();
    }
    if (threadIdx.x == 0) pmax[blockIdx.x] = s[0];
}

__global__ __launch_bounds__(256)
void k_max_fin(const float* __restrict__ pmax, float* __restrict__ scal, int p) {
    __shared__ float s[256];
    float m = NEG_INF_F;
    for (int j = threadIdx.x; j < p; j += 256) m = fmaxf(m, pmax[j]);
    s[threadIdx.x] = m;
    __syncthreads();
    for (int st = 128; st > 0; st >>= 1) {
        if (threadIdx.x < st) s[threadIdx.x] = fmaxf(s[threadIdx.x], s[threadIdx.x + st]);
        __syncthreads();
    }
    if (threadIdx.x == 0) scal[0] = fmaxf(s[0], scal[2]);   // include 'do nothing'
}

__global__ __launch_bounds__(256)
void k_sum_part(const float* __restrict__ agg5, const float* __restrict__ b5,
                const int* __restrict__ ready, const float* __restrict__ scal,
                float* __restrict__ psum, int n) {
    __shared__ float s[256];
    int i = blockIdx.x * 256 + threadIdx.x;
    float gmax = scal[0];
    float v = 0.0f;
    if (i < n && ready[i] != 0) v = __expf(agg5[i] + b5[0] - gmax);
    s[threadIdx.x] = v;
    __syncthreads();
    for (int st = 128; st > 0; st >>= 1) {
        if (threadIdx.x < st) s[threadIdx.x] += s[threadIdx.x + st];
        __syncthreads();
    }
    if (threadIdx.x == 0) psum[blockIdx.x] = s[0];
}

__global__ __launch_bounds__(256)
void k_sum_fin(const float* __restrict__ psum, float* __restrict__ scal, int p) {
    __shared__ float s[256];
    float v = 0.0f;
    for (int j = threadIdx.x; j < p; j += 256) v += psum[j];
    s[threadIdx.x] = v;
    __syncthreads();
    for (int st = 128; st > 0; st >>= 1) {
        if (threadIdx.x < st) s[threadIdx.x] += s[threadIdx.x + st];
        __syncthreads();
    }
    if (threadIdx.x == 0) scal[1] = s[0] + __expf(scal[2] - scal[0]);
}

__global__ __launch_bounds__(256)
void k_write_probs(const float* __restrict__ agg5, const float* __restrict__ b5,
                   const int* __restrict__ ready, const float* __restrict__ scal,
                   float* __restrict__ out, int n) {
    int i = blockIdx.x * 256 + threadIdx.x;
    if (i > n) return;
    float gmax = scal[0], inv = 1.0f / scal[1];
    if (i == n) {
        out[n] = __expf(scal[2] - gmax) * inv;
    } else {
        out[i] = (ready[i] != 0) ? __expf(agg5[i] + b5[0] - gmax) * inv : 0.0f;
    }
}

// ------------------------------- launcher ----------------------------------

extern "C" void kernel_launch(void* const* d_in, const int* in_sizes, int n_in,
                              void* d_out, int out_size, void* d_ws, size_t ws_size,
                              hipStream_t stream) {
    const float* x    = (const float*)d_in[0];
    const int*   ei   = (const int*)d_in[1];
    const int*   rdy  = (const int*)d_in[2];
    const float* W1   = (const float*)d_in[3];
    const float* b1   = (const float*)d_in[4];
    const float* W2   = (const float*)d_in[5];
    const float* b2   = (const float*)d_in[6];
    const float* W3   = (const float*)d_in[7];
    const float* b3   = (const float*)d_in[8];
    const float* W4   = (const float*)d_in[9];
    const float* b4   = (const float*)d_in[10];
    const float* W5   = (const float*)d_in[11];
    const float* b5   = (const float*)d_in[12];
    const float* Wdn  = (const float*)d_in[13];
    const float* bdn  = (const float*)d_in[14];
    const float* Wv   = (const float*)d_in[15];
    const float* bv   = (const float*)d_in[16];
    float* out = (float*)d_out;

    const int N = in_sizes[2];          // ready is [N,1]
    const int E = in_sizes[1] / 2;      // edge_index is [2,E]
    const int DIN = 128, H = 64;
    const int P = (N + 255) / 256;

    const int* rowp = ei;
    const int* colp = ei + E;

    // workspace carve-up (256B aligned blocks)
    char* ws = (char*)d_ws;
    auto carve = [&](size_t bytes) -> void* {
        void* p = (void*)ws;
        ws += (bytes + 255) & ~(size_t)255;
        return p;
    };
    float*    dinv = (float*)   carve((size_t)N * 4);          // deg -> dinv in place
    uint16_t* xb   = (uint16_t*)carve((size_t)N * DIN * 2);
    uint16_t* hb   = (uint16_t*)carve((size_t)N * H * 2);
    uint16_t* WT   = (uint16_t*)carve((size_t)H * DIN * 2);
    float*    t    = (float*)   carve((size_t)N * H * 4);
    float*    agg  = (float*)   carve((size_t)N * H * 4);
    float*    h    = (float*)   carve((size_t)N * H * 4);
    float*    t5   = (float*)   carve((size_t)N * 4);
    float*    agg5 = (float*)   carve((size_t)N * 4);
    float*    xsum = (float*)   carve(64 * 4);
    float*    pmax = (float*)   carve((size_t)P * 4);
    float*    psum = (float*)   carve((size_t)P * 4);
    float*    scal = (float*)   carve(8 * 4);

    const int T = 256;
    dim3 blk(T);
    auto g1 = [&](long n) { return dim3((unsigned)((n + T - 1) / T)); };

    // ---- normalization: deg (with self loop) -> dinv
    k_init<<<g1(N), blk, 0, stream>>>(dinv, xsum, N);
    k_deg_edges<<<g1(E), blk, 0, stream>>>(rowp, dinv, E);
    k_rsqrt_inplace<<<g1(N), blk, 0, stream>>>(dinv, N);

    // ---- input to bf16
    k_convert_bf16<<<g1((long)N * DIN), blk, 0, stream>>>(x, xb, N * DIN);

    const dim3 gemmBlk(128);
    const dim3 gemmGrd((unsigned)((N + 63) / 64));
    const dim3 edgeGrd((unsigned)((E + 7) / 8));

    // ---- layer 1 (K = 128)
    k_prep_wt<<<g1(DIN * H), blk, 0, stream>>>(W1, WT, DIN);
    k_gemm<128><<<gemmGrd, gemmBlk, 0, stream>>>(xb, WT, t, N);
    k_selfinit<<<g1((long)N * H), blk, 0, stream>>>(dinv, t, agg, N * H);
    k_edge_agg<<<edgeGrd, blk, 0, stream>>>(rowp, colp, dinv, t, agg, E);
    k_relu_bias<<<g1((long)N * H), blk, 0, stream>>>(agg, b1, h, hb, N * H);

    // ---- layers 2..4 (K = 64)
    const float* Ws[3] = {W2, W3, W4};
    const float* bs[3] = {b2, b3, b4};
    for (int l = 0; l < 3; ++l) {
        k_prep_wt<<<g1(H * H), blk, 0, stream>>>(Ws[l], WT, H);
        k_gemm<64><<<gemmGrd, gemmBlk, 0, stream>>>(hb, WT, t, N);
        k_selfinit<<<g1((long)N * H), blk, 0, stream>>>(dinv, t, agg, N * H);
        k_edge_agg<<<edgeGrd, blk, 0, stream>>>(rowp, colp, dinv, t, agg, E);
        k_relu_bias<<<g1((long)N * H), blk, 0, stream>>>(agg, bs[l], h, hb, N * H);
    }

    // ---- layer 5 (scalar conv) + heads
    k_dot64<<<g1(N), blk, 0, stream>>>(h, W5, t5, N);
    k_selfinit_s<<<g1(N), blk, 0, stream>>>(dinv, t5, agg5, N);
    k_edge_agg_s<<<g1(E), blk, 0, stream>>>(rowp, colp, dinv, t5, agg5, E);
    k_xmean_part<<<dim3((unsigned)((N + 511) / 512)), dim3(64), 0, stream>>>(h, xsum, N);
    k_head<<<dim3(1), dim3(64), 0, stream>>>(xsum, Wv, bv, Wdn, bdn, out, scal, N);

    // ---- masked softmax over [N+1] logits
    k_max_part<<<dim3((unsigned)P), blk, 0, stream>>>(agg5, b5, rdy, pmax, N);
    k_max_fin<<<dim3(1), blk, 0, stream>>>(pmax, scal, P);
    k_sum_part<<<dim3((unsigned)P), blk, 0, stream>>>(agg5, b5, rdy, scal, psum, N);
    k_sum_fin<<<dim3(1), blk, 0, stream>>>(psum, scal, P);
    k_write_probs<<<g1((long)N + 1), blk, 0, stream>>>(agg5, b5, rdy, scal, out, N);
}
